// GeneralSheafs_2594160246967
// MI455X (gfx1250) — matrix operand order repaired and checked by reading, not compiled, not run on hardware
//
#include <hip/hip_runtime.h>
#include <hip/hip_bf16.h>
#include <math.h>

// ---------------------------------------------------------------------------
// Problem constants (match reference)
// ---------------------------------------------------------------------------
#define Nn    10000
#define Ee    5000
#define NNZ   160000
#define F_IN  64
#define Dd    4
#define Hh    64
#define Cc    40
#define Ndx   (Nn * Dd)   // 40000
#define Edx   (Ee * Dd)   // 20000

typedef __attribute__((ext_vector_type(16))) _Float16 v16h;
typedef __attribute__((ext_vector_type(8)))  float    v8f;

// ---------------------------------------------------------------------------
// f32 -> f16 convert (activations)
// ---------------------------------------------------------------------------
__global__ void k_cvt_f16(const float* __restrict__ src, _Float16* __restrict__ dst, int n) {
  int i = blockIdx.x * blockDim.x + threadIdx.x;
  if (i < n) dst[i] = (_Float16)src[i];
}

// f32 (K x Nc, row-major) -> f16 transposed (Nc x K, row-major)
__global__ void k_cvt_tr_f16(const float* __restrict__ src, _Float16* __restrict__ dst,
                             int K, int Nc) {
  int i = blockIdx.x * blockDim.x + threadIdx.x;
  if (i < K * Nc) {
    int k = i / Nc, n = i - k * Nc;
    dst[n * K + k] = (_Float16)src[i];
  }
}

// ---------------------------------------------------------------------------
// WMMA fragment loaders (wave32; A and B both use contiguous k order, not the ISA table order)
//  A 16x32 f16 : lane l -> row m=l&15; lanes<16: K=[k0..k0+15]
//                lanes>=16: K=[k0+16..k0+31]  (32 contiguous bytes)
//  B 32x16 f16 : lane l -> col n=l&15; lanes<16: K=[k0..k0+15], lanes>=16: +16
//                (from PRE-TRANSPOSED Bt[Nc x K] -> 32 contiguous bytes)
//  C/D 16x16 f32: lane l -> col n=l&15; rows m=(l<16?0:8)+r, r=0..7
// ---------------------------------------------------------------------------
__device__ __forceinline__ v16h load_a_frag(const _Float16* __restrict__ p /* row base + k0 */,
                                            int lane) {
  const _Float16* q = p + (lane >> 4) * 16;
  v16h a;
#pragma unroll
  for (int j = 0; j < 8; ++j) a[j] = q[j];
#pragma unroll
  for (int j = 0; j < 8; ++j) a[8 + j] = q[8 + j];
  return a;
}

template <int K>
__device__ __forceinline__ v16h load_bt_frag(const _Float16* __restrict__ Bt,
                                             int n, int k0, int lane) {
  const _Float16* p = Bt + n * K + k0 + ((lane >> 4) << 4);
  v16h b;
#pragma unroll
  for (int j = 0; j < 16; ++j) b[j] = p[j];
  return b;
}

// Fast sigmoid: raw v_rcp_f32 (feeds an f16 pipeline; IEEE div not needed)
__device__ __forceinline__ float fast_sigmoid(float x) {
  return __builtin_amdgcn_rcpf(1.0f + __expf(-x));
}

// ---------------------------------------------------------------------------
// Templated WMMA GEMM: C(MxNc) = A(MxK) @ B(KxNc), B given transposed (NcxK).
// One wave per (16 x 16*NT) output strip: A fragment reused across NT n-tiles.
// Fully unrolled K loop; compile-time masks; 32-bit addressing.
// ---------------------------------------------------------------------------
template <int K, int Nc, int NT, bool WF32, bool WF16, bool MASK_M>
__global__ void k_wmma_gemm(const _Float16* __restrict__ A, const _Float16* __restrict__ Bt,
                            float* __restrict__ Cf, _Float16* __restrict__ Ch, int M) {
  constexpr bool MASK_N = (Nc % 16) != 0;
  const int lane   = threadIdx.x;
  const int m_tile = blockIdx.x * 16;
  const int n_base = blockIdx.y * (16 * NT);

  int m_a = m_tile + (lane & 15);
  if (MASK_M && m_a >= M) m_a = M - 1;       // clamp A row; stores masked
  const _Float16* Arow = A + m_a * K;

  v8f acc[NT];
#pragma unroll
  for (int t = 0; t < NT; ++t) acc[t] = (v8f){};

#pragma unroll
  for (int k0 = 0; k0 < K; k0 += 32) {
    v16h a = load_a_frag(Arow + k0, lane);
#pragma unroll
    for (int t = 0; t < NT; ++t) {
      int n = n_base + t * 16 + (lane & 15);
      if (MASK_N && n >= Nc) n = Nc - 1;
      v16h b = load_bt_frag<K>(Bt, n, k0, lane);
      acc[t] = __builtin_amdgcn_wmma_f32_16x16x32_f16(false, a, false, b,
                                                      (short)0, acc[t], false, false);
    }
  }

  const int m_off = (lane >> 4) * 8;
#pragma unroll
  for (int t = 0; t < NT; ++t) {
    const int n = n_base + t * 16 + (lane & 15);
    if (MASK_N && n >= Nc) continue;
#pragma unroll
    for (int r = 0; r < 8; ++r) {
      const int m = m_tile + m_off + r;
      if (MASK_M && m >= M) continue;
      if (WF32) Cf[m * Nc + n] = acc[t][r];
      if (WF16) Ch[m * Nc + n] = (_Float16)acc[t][r];
    }
  }
}

// ---------------------------------------------------------------------------
// Sheaf block GEMM: h[i,0:16] = sigmoid( concat(xN[row[i]], eE[col[i]]) @ W )
// 16 incidences per wave; K=128 (first 64 from xN, last 64 from eE); Nc=16.
// W pre-transposed: Wt is 16 x 128.
// ---------------------------------------------------------------------------
__global__ void k_sheaf(const _Float16* __restrict__ xNh, const _Float16* __restrict__ eEh,
                        const _Float16* __restrict__ Wt,
                        const int* __restrict__ row, const int* __restrict__ col,
                        float* __restrict__ hout) {
  const int lane = threadIdx.x;
  const int tile = blockIdx.x;
  const int m    = tile * 16 + (lane & 15);   // NNZ % 16 == 0: no mask needed
  const int r    = row[m];
  const int c    = col[m];
  const int n    = lane & 15;

  const _Float16* xrow = xNh + r * 64;
  const _Float16* erow = eEh + c * 64;

  v8f acc = {};
#pragma unroll
  for (int k0 = 0; k0 < 128; k0 += 32) {
    const _Float16* src = (k0 < 64) ? (xrow + k0) : (erow + (k0 - 64));
    v16h a = load_a_frag(src, lane);
    v16h b = load_bt_frag<128>(Wt, n, k0, lane);
    acc = __builtin_amdgcn_wmma_f32_16x16x32_f16(false, a, false, b, (short)0, acc, false, false);
  }

  const int m_off = (lane >> 4) * 8;
#pragma unroll
  for (int rr = 0; rr < 8; ++rr) {
    int mm = tile * 16 + m_off + rr;
    hout[mm * 16 + n] = fast_sigmoid(acc[rr]);
  }
}

// ---------------------------------------------------------------------------
// Degrees: node_deg[row]++, edge_deg[col]++   (expanded deg = 4*these)
// ---------------------------------------------------------------------------
__global__ void k_degree(const int* __restrict__ row, const int* __restrict__ col,
                         float* __restrict__ ndeg, float* __restrict__ edeg, int nnz) {
  int i = blockIdx.x * blockDim.x + threadIdx.x;
  if (i < nnz) {
    atomicAdd(&ndeg[row[i]], 1.0f);
    atomicAdd(&edeg[col[i]], 1.0f);
  }
}

// In-place: p = (p>0) ? 1/(4p) : 0   (expanded-space inverse degree)
__global__ void k_invert(float* __restrict__ p, int n) {
  int i = blockIdx.x * blockDim.x + threadIdx.x;
  if (i < n) { float v = p[i]; p[i] = (v > 0.0f) ? 0.25f / v : 0.0f; }
}

// ---------------------------------------------------------------------------
// Node -> edge scatter: acc_e[4c+j, :] += sum_i H[i,j] * xb[4r+i, :]
// One block (64 lanes = feature dim) per incidence. H address is uniform
// per block -> scalar loads. L2-atomic-bound by design.
// ---------------------------------------------------------------------------
__global__ void k_scatter_ne(const float* __restrict__ h, const int* __restrict__ row,
                             const int* __restrict__ col, const float* __restrict__ xb,
                             float* __restrict__ acc_e) {
  const int i = blockIdx.x;
  const int t = threadIdx.x;                  // 0..63
  const int r = row[i], c = col[i];
  const float* hb = h + i * 16;
  const float* xr = xb + (4 * r) * 64 + t;
  float x0 = xr[0], x1 = xr[64], x2 = xr[128], x3 = xr[192];
  float* er = acc_e + (4 * c) * 64 + t;
#pragma unroll
  for (int j = 0; j < 4; ++j) {
    float y = hb[0 + j] * x0 + hb[4 + j] * x1 + hb[8 + j] * x2 + hb[12 + j] * x3;
    atomicAdd(er + j * 64, y);
  }
}

// acc_e[idx] *= Binv[edge];  edge = idx / (4*64)
__global__ void k_norm_e(float* __restrict__ acc_e, const float* __restrict__ Binv, int n) {
  int i = blockIdx.x * blockDim.x + threadIdx.x;
  if (i < n) acc_e[i] *= Binv[i >> 8];
}

// Edge -> node scatter: acc_n[4r+i, :] += sum_j H[i,j] * acc_e[4c+j, :]
__global__ void k_scatter_en(const float* __restrict__ h, const int* __restrict__ row,
                             const int* __restrict__ col, const float* __restrict__ acc_e,
                             float* __restrict__ acc_n) {
  const int i = blockIdx.x;
  const int t = threadIdx.x;
  const int r = row[i], c = col[i];
  const float* hb = h + i * 16;
  const float* gr = acc_e + (4 * c) * 64 + t;
  float g0 = gr[0], g1 = gr[64], g2 = gr[128], g3 = gr[192];
  float* nr = acc_n + (4 * r) * 64 + t;
#pragma unroll
  for (int j = 0; j < 4; ++j) {
    float z = hb[4 * j + 0] * g0 + hb[4 * j + 1] * g1 + hb[4 * j + 2] * g2 + hb[4 * j + 3] * g3;
    atomicAdd(nr + j * 64, z);
  }
}

// dst_h16[idx] = (elu?)( Dinv[node] * acc_n[idx] );  node = idx >> 8
__global__ void k_post(const float* __restrict__ acc_n, const float* __restrict__ Dinv,
                       _Float16* __restrict__ dst, int n, int do_elu) {
  int i = blockIdx.x * blockDim.x + threadIdx.x;
  if (i < n) {
    float v = Dinv[i >> 8] * acc_n[i];
    if (do_elu) v = (v > 0.0f) ? v : (__expf(v) - 1.0f);
    dst[i] = (_Float16)v;
  }
}

// ---------------------------------------------------------------------------
// Host-side orchestration
// ---------------------------------------------------------------------------
static inline int cdiv(int a, int b) { return (a + b - 1) / b; }

extern "C" void kernel_launch(void* const* d_in, const int* in_sizes, int n_in,
                              void* d_out, int out_size, void* d_ws, size_t ws_size,
                              hipStream_t stream) {
  const float* x    = (const float*)d_in[0];
  const float* he   = (const float*)d_in[1];
  const int*   row  = (const int*)d_in[2];
  const int*   col  = row + NNZ;
  const float* linW = (const float*)d_in[3];
  const float* shW  = (const float*)d_in[4];
  const float* c0W  = (const float*)d_in[5];
  const float* c1W  = (const float*)d_in[6];
  const float* l2W  = (const float*)d_in[7];
  float* out = (float*)d_out;

  // --- workspace carve-out (~56 MB total; resident in 192MB L2) ---
  char* w = (char*)d_ws;
  size_t off = 0;
  auto alloc = [&](size_t bytes) -> void* {
    void* p = (void*)(w + off);
    off = (off + bytes + 255) & ~(size_t)255;
    return p;
  };
  _Float16* xh    = (_Float16*)alloc((size_t)Nn * F_IN * 2);
  _Float16* ehh   = (_Float16*)alloc((size_t)Ee * F_IN * 2);
  _Float16* linWt = (_Float16*)alloc((size_t)256 * 64 * 2);    // 256 x 64 (transposed)
  _Float16* shWt  = (_Float16*)alloc((size_t)16 * 128 * 2);    // 16 x 128
  _Float16* c0Wt  = (_Float16*)alloc((size_t)64 * 64 * 2);     // 64 x 64
  _Float16* c1Wt  = (_Float16*)alloc((size_t)64 * 64 * 2);
  _Float16* l2Wt  = (_Float16*)alloc((size_t)40 * 256 * 2);    // 40 x 256
  _Float16* xNh   = (_Float16*)alloc((size_t)Ndx * 64 * 2);
  _Float16* eEh   = (_Float16*)alloc((size_t)Edx * 64 * 2);
  float*    hbuf  = (float*)alloc((size_t)NNZ * 16 * 4);
  float*    ndeg  = (float*)alloc((size_t)Nn * 4);   // becomes Dinv
  float*    edeg  = (float*)alloc((size_t)Ee * 4);   // becomes Binv
  float*    xb    = (float*)alloc((size_t)Ndx * 64 * 4);
  float*    acc_e = (float*)alloc((size_t)Edx * 64 * 4);
  float*    acc_n = (float*)alloc((size_t)Ndx * 64 * 4);
  _Float16* h1h   = (_Float16*)alloc((size_t)Ndx * 64 * 2);
  _Float16* h2h   = (_Float16*)alloc((size_t)Ndx * 64 * 2);
  (void)ws_size; (void)n_in; (void)in_sizes; (void)out_size;

  // --- convert activations to f16, weights to f16-transposed ---
  k_cvt_f16<<<cdiv(Nn * F_IN, 256), 256, 0, stream>>>(x, xh, Nn * F_IN);
  k_cvt_f16<<<cdiv(Ee * F_IN, 256), 256, 0, stream>>>(he, ehh, Ee * F_IN);
  k_cvt_tr_f16<<<cdiv(64 * 256, 256), 256, 0, stream>>>(linW, linWt, 64, 256);
  k_cvt_tr_f16<<<cdiv(128 * 16, 256), 256, 0, stream>>>(shW, shWt, 128, 16);
  k_cvt_tr_f16<<<cdiv(64 * 64, 256), 256, 0, stream>>>(c0W, c0Wt, 64, 64);
  k_cvt_tr_f16<<<cdiv(64 * 64, 256), 256, 0, stream>>>(c1W, c1Wt, 64, 64);
  k_cvt_tr_f16<<<cdiv(256 * 40, 256), 256, 0, stream>>>(l2W, l2Wt, 256, 40);

  // --- lin: xN = x@lin_W (reshape free), eE = he@lin_W ---
  { dim3 g(Nn / 16, 256 / 64);           // K=64, Nc=256, NT=4
    k_wmma_gemm<64, 256, 4, false, true, false>
        <<<g, 32, 0, stream>>>(xh, linWt, nullptr, xNh, Nn); }
  { dim3 g(cdiv(Ee, 16), 256 / 64);      // M=5000 ragged
    k_wmma_gemm<64, 256, 4, false, true, true>
        <<<g, 32, 0, stream>>>(ehh, linWt, nullptr, eEh, Ee); }

  // --- sheaf blocks: h = sigmoid(concat(xN[row], eE[col]) @ sheaf_W) ---
  k_sheaf<<<NNZ / 16, 32, 0, stream>>>(xNh, eEh, shWt, row, col, hbuf);

  // --- degrees (shared by both conv layers) ---
  hipMemsetAsync(ndeg, 0, (size_t)Nn * 4, stream);
  hipMemsetAsync(edeg, 0, (size_t)Ee * 4, stream);
  k_degree<<<cdiv(NNZ, 256), 256, 0, stream>>>(row, col, ndeg, edeg, NNZ);
  k_invert<<<cdiv(Nn, 256), 256, 0, stream>>>(ndeg, Nn);
  k_invert<<<cdiv(Ee, 256), 256, 0, stream>>>(edeg, Ee);

  // --- conv layer 0 : h1 = elu( D^-1 Hs ( B^-1 Hs^T (xN @ W0) ) ) ---
  { dim3 g(Ndx / 16, 1);                 // K=64, Nc=64, NT=4
    k_wmma_gemm<64, 64, 4, true, false, false>
        <<<g, 32, 0, stream>>>(xNh, c0Wt, xb, nullptr, Ndx); }
  hipMemsetAsync(acc_e, 0, (size_t)Edx * 64 * 4, stream);
  k_scatter_ne<<<NNZ, 64, 0, stream>>>(hbuf, row, col, xb, acc_e);
  k_norm_e<<<cdiv(Edx * 64, 256), 256, 0, stream>>>(acc_e, edeg, Edx * 64);
  hipMemsetAsync(acc_n, 0, (size_t)Ndx * 64 * 4, stream);
  k_scatter_en<<<NNZ, 64, 0, stream>>>(hbuf, row, col, acc_e, acc_n);
  k_post<<<cdiv(Ndx * 64, 256), 256, 0, stream>>>(acc_n, ndeg, h1h, Ndx * 64, 1);

  // --- conv layer 1 : h2 = D^-1 Hs ( B^-1 Hs^T (h1 @ W1) ) ---
  { dim3 g(Ndx / 16, 1);
    k_wmma_gemm<64, 64, 4, true, false, false>
        <<<g, 32, 0, stream>>>(h1h, c1Wt, xb, nullptr, Ndx); }
  hipMemsetAsync(acc_e, 0, (size_t)Edx * 64 * 4, stream);
  k_scatter_ne<<<NNZ, 64, 0, stream>>>(hbuf, row, col, xb, acc_e);
  k_norm_e<<<cdiv(Edx * 64, 256), 256, 0, stream>>>(acc_e, edeg, Edx * 64);
  hipMemsetAsync(acc_n, 0, (size_t)Ndx * 64 * 4, stream);
  k_scatter_en<<<NNZ, 64, 0, stream>>>(hbuf, row, col, acc_e, acc_n);
  k_post<<<cdiv(Ndx * 64, 256), 256, 0, stream>>>(acc_n, ndeg, h2h, Ndx * 64, 0);

  // --- final: out(10000x40) = h2.reshape(N,256) @ lin2_W ---
  { dim3 g(Nn / 16, 1);                  // K=256, Nc=40, NT=3 (n-masked)
    k_wmma_gemm<256, 40, 3, true, false, false>
        <<<g, 32, 0, stream>>>(h2h, l2Wt, out, nullptr, Nn); }
}
